// GRU_70763881168990
// MI455X (gfx1250) — matrix operand order repaired and checked by reading, not compiled, not run on hardware
//
#include <hip/hip_runtime.h>
#include <hip/hip_bf16.h>

typedef __attribute__((ext_vector_type(16))) __bf16 v16bf;
typedef __attribute__((ext_vector_type(8)))  __bf16 v8bf;
typedef __attribute__((ext_vector_type(8)))  float  v8f;

namespace {
constexpr int kBatch = 512;
constexpr int kL     = 1024;
constexpr int kW     = 256;   // hidden width
constexpr int kG     = 768;   // 3*W
constexpr int kBT    = 16;    // batch rows per workgroup (1 WMMA M-tile)
constexpr int kWaves = 16;
constexpr int kTPB   = kWaves * 32;
}

// ---------------- pre-kernel: w_hh fp32 -> bf16 into workspace ----------------
__global__ __launch_bounds__(256) void gru_cvt_whh(const float* __restrict__ w,
                                                   __bf16* __restrict__ o, int n) {
  int i = blockIdx.x * blockDim.x + threadIdx.x;
  if (i < n) o[i] = (__bf16)w[i];
}

// A-fragment (16x32 bf16): lane m = lane&15, K = {kb..kb+15}, kb=(lane>>4)*16
__device__ __forceinline__ v16bf load_a_frag(const __bf16* hrow, int c, int kb) {
  v8bf lo = *(const v8bf*)(hrow + c * 32 + kb);
  v8bf hi = *(const v8bf*)(hrow + c * 32 + kb + 8);
  v16bf a;
#pragma unroll
  for (int j = 0; j < 8; ++j) { a[j] = lo[j]; a[8 + j] = hi[j]; }
  return a;
}

__device__ __forceinline__ v8f wmma_acc(v16bf a, v16bf b, v8f c) {
  return __builtin_amdgcn_wmma_f32_16x16x32_bf16(false, a, false, b, (short)0, c,
                                                 false, false);
}

__global__ __launch_bounds__(kTPB) __attribute__((amdgpu_waves_per_eu(4, 4)))
void gru_scan(
    const float* __restrict__ x,     // [B, L]
    const float* __restrict__ w_ih,  // [768, 2]
    const __bf16* __restrict__ whh,  // [768, 256] bf16 (workspace)
    const float* __restrict__ b_ih,  // [768]
    const float* __restrict__ b_hh,  // [768]
    const float* __restrict__ fc_w,  // [2, 256]
    const float* __restrict__ fc_b,  // [2]
    float* __restrict__ out) {       // [B]
  __shared__ __bf16 hbuf[2][kBT][kW];   // double-buffered hidden state (bf16)
  __shared__ float  gi3[3][kG];         // 3-way input-gate table (incl b_ih)
  __shared__ float  bhh[kG];
  __shared__ __bf16 fcw[2][kW];
  __shared__ float  fcb[2];
  __shared__ float  lpacc[kBT];         // per-row log-prob accumulators
  __shared__ unsigned int selt[kL + 1]; // 2 bits per batch row per step

  const int tid  = threadIdx.x;
  const int wave = tid >> 5;
  const int lane = tid & 31;
  const int nlo  = lane & 15;
  const int half = lane >> 4;
  const int b0   = blockIdx.x * kBT;

  // ---------------- one-time setup ----------------
  for (int g = tid; g < kG; g += kTPB) {
    float w0 = w_ih[2 * g], w1 = w_ih[2 * g + 1], bi = b_ih[g];
    gi3[0][g] = bi + w0;                 // x_prev = +1 -> emb [1,0]
    gi3[1][g] = bi + w1;                 // x_prev = -1 -> emb [0,1]
    gi3[2][g] = bi + 0.5f * (w0 + w1);   // x_init = 0  -> emb [0.5,0.5]
    bhh[g] = b_hh[g];
  }
  for (int i = tid; i < 2 * kW; i += kTPB) fcw[i >> 8][i & (kW - 1)] = (__bf16)fc_w[i];
  if (tid < 2) fcb[tid] = fc_b[tid];
  if (tid < kBT) lpacc[tid] = 0.f;
  for (int i = tid; i < kBT * kW; i += kTPB) ((__bf16*)hbuf)[i] = (__bf16)0.f;  // h0 = 0
  for (int t = tid; t <= kL; t += kTPB) {
    unsigned int sv;
    if (t == 0) {
      sv = 0xAAAAAAAAu;  // code 2 (x_init) for all 16 rows
    } else {
      sv = 0;
      for (int m = 0; m < kBT; ++m) {
        unsigned int code = (x[(size_t)(b0 + m) * kL + (t - 1)] > 0.f) ? 0u : 1u;
        sv |= code << (2 * m);
      }
    }
    selt[t] = sv;
  }
  __syncthreads();

  const int hw = wave * 16 + nlo;  // hidden unit owned by this lane (all 3 gates)
  const int kb = half * 16;        // A-frag K sub-base
  const int bk = half * 16;        // B-frag K sub-base
  const float bv0 = bhh[hw], bv1 = bhh[kW + hw], bv2 = bhh[2 * kW + hw];
  const float fb0 = fcb[0], fb1 = fcb[1];

  float hold[8];
#pragma unroll
  for (int r = 0; r < 8; ++r) hold[r] = 0.f;

  const __bf16* wr0 = whh + (size_t)hw * kW;           // r-gate weight row
  const __bf16* wr1 = wr0 + (size_t)kW * kW;           // z-gate
  const __bf16* wr2 = wr1 + (size_t)kW * kW;           // n-gate (streamed from L2)

  // r/z-gate weight slices pinned in registers for the whole scan (128 VGPRs)
  v16bf wR[8], wZ[8];
#pragma unroll
  for (int c = 0; c < 8; ++c) {
    wR[c] = *(const v16bf*)(wr0 + c * 32 + bk);
    wZ[c] = *(const v16bf*)(wr1 + c * 32 + bk);
  }

  for (int t = 0; t < kL; ++t) {
    const __bf16* hin = &hbuf[t & 1][0][0];
    __bf16* hout = &hbuf[(t + 1) & 1][0][0];

    // Opacify the lane-divergent ELEMENT OFFSET (not the pointer) so LICM
    // cannot hoist the n-gate fragment loads out of the 1024-step loop, while
    // the pointer chain still roots at the global kernel arg -> addrspace(1)
    // is inferable and the loads lower to global_load_b128 (LOADcnt only,
    // no flat-path DScnt coupling with the LDS A-fragment loads).
    unsigned noff = (unsigned)bk;
    asm("" : "+v"(noff));
    const __bf16* wr2t = wr2 + noff;

    v8f acc0, acc1, acc2;
#pragma unroll
    for (int r = 0; r < 8; ++r) { acc0[r] = bv0; acc1[r] = bv1; acc2[r] = bv2; }

#pragma unroll
    for (int c = 0; c < 8; ++c) {
      v16bf a = load_a_frag(hin + (size_t)nlo * kW, c, kb);
      v16bf bN = *(const v16bf*)(wr2t + c * 32);  // L2-resident stream
      acc0 = wmma_acc(a, wR[c], acc0);
      acc1 = wmma_acc(a, wZ[c], acc1);
      acc2 = wmma_acc(a, bN, acc2);
    }

    const unsigned int sv = selt[t];
#pragma unroll
    for (int r = 0; r < 8; ++r) {
      int m = half * 8 + r;
      int sc = (int)((sv >> (2 * m)) & 3u);
      float gr = gi3[sc][hw], gz = gi3[sc][kW + hw], gn = gi3[sc][2 * kW + hw];
      float rg = 1.f / (1.f + __expf(-(gr + acc0[r])));
      float zg = 1.f / (1.f + __expf(-(gz + acc1[r])));
      float pre = gn + rg * acc2[r];
      float e2 = __expf(2.f * pre);
      float ng = (e2 - 1.f) / (e2 + 1.f);        // tanh
      float hn = (1.f - zg) * ng + zg * hold[r];
      hold[r] = hn;
      hout[(size_t)m * kW + hw] = (__bf16)hn;
    }

    // log-softmax head on h_t -> lp_{t-1}; runs after acc registers are dead
    if (wave == 0 && t > 0) {  // wave-uniform branch: EXEC all-1 at WMMA
      v8f accF;
#pragma unroll
      for (int r = 0; r < 8; ++r) accF[r] = 0.f;
#pragma unroll
      for (int c = 0; c < 8; ++c) {
        v16bf a = load_a_frag(hin + (size_t)nlo * kW, c, kb);
        v16bf bF;
        if (nlo < 2) {
          bF = *(const v16bf*)(&fcw[nlo][c * 32 + bk]);
        } else {
#pragma unroll
          for (int j = 0; j < 16; ++j) bF[j] = (__bf16)0.f;
        }
        accF = wmma_acc(a, bF, accF);
      }
#pragma unroll
      for (int r = 0; r < 8; ++r) {
        float v0 = accF[r] + fb0;
        float v1 = __shfl(accF[r] + fb1, lane + 1, 32);  // logit1 -> lanes 0,16
        if (nlo == 0) {
          int m = half * 8 + r;
          float mx = fmaxf(v0, v1);
          float lse = mx + __logf(__expf(v0 - mx) + __expf(v1 - mx));
          int msk = (int)((sv >> (2 * m)) & 3u);  // sel[t] == sign of x[.,t-1]
          lpacc[m] += (msk == 0) ? (v0 - lse) : (v1 - lse);
        }
      }
    }
    __syncthreads();
  }

  // ---------------- final step's log-prob on h_L ----------------
  if (wave == 0) {
    const __bf16* hin = &hbuf[kL & 1][0][0];
    v8f accF;
#pragma unroll
    for (int r = 0; r < 8; ++r) accF[r] = 0.f;
#pragma unroll
    for (int c = 0; c < 8; ++c) {
      v16bf a = load_a_frag(hin + (size_t)nlo * kW, c, kb);
      v16bf bF;
      if (nlo < 2) {
        bF = *(const v16bf*)(&fcw[nlo][c * 32 + bk]);
      } else {
#pragma unroll
        for (int j = 0; j < 16; ++j) bF[j] = (__bf16)0.f;
      }
      accF = wmma_acc(a, bF, accF);
    }
    const unsigned int sv = selt[kL];
#pragma unroll
    for (int r = 0; r < 8; ++r) {
      float v0 = accF[r] + fb0;
      float v1 = __shfl(accF[r] + fb1, lane + 1, 32);
      if (nlo == 0) {
        int m = half * 8 + r;
        float mx = fmaxf(v0, v1);
        float lse = mx + __logf(__expf(v0 - mx) + __expf(v1 - mx));
        int msk = (int)((sv >> (2 * m)) & 3u);
        out[b0 + m] = lpacc[m] + ((msk == 0) ? (v0 - lse) : (v1 - lse));
      }
    }
  }
}

extern "C" void kernel_launch(void* const* d_in, const int* in_sizes, int n_in,
                              void* d_out, int out_size, void* d_ws, size_t ws_size,
                              hipStream_t stream) {
  const float* x    = (const float*)d_in[0];
  const float* w_ih = (const float*)d_in[1];
  const float* w_hh = (const float*)d_in[2];
  const float* b_ih = (const float*)d_in[3];
  const float* b_hh = (const float*)d_in[4];
  const float* fc_w = (const float*)d_in[5];
  const float* fc_b = (const float*)d_in[6];

  __bf16* whh_bf = (__bf16*)d_ws;          // 768*256*2 = 384 KB of workspace
  const int nW = kG * kW;
  gru_cvt_whh<<<(nW + 255) / 256, 256, 0, stream>>>(w_hh, whh_bf, nW);

  gru_scan<<<kBatch / kBT, kTPB, 0, stream>>>(x, w_ih, whh_bf, b_ih, b_hh,
                                              fc_w, fc_b, (float*)d_out);
}
